// TriangleMultiplication_74990128988755
// MI455X (gfx1250) — matrix operand order repaired and checked by reading, not compiled, not run on hardware
//
#include <hip/hip_runtime.h>
#include <stdint.h>

#define N_DIM 768
#define C_DIM 128
#define NN    (768*768)

typedef __attribute__((ext_vector_type(16))) __bf16 v16bf;
typedef __attribute__((ext_vector_type(2)))  __bf16 v2bf;
typedef __attribute__((ext_vector_type(8)))  float  v8f;

union FragA { uint4 q[2]; v16bf v; };

__device__ __forceinline__ unsigned short f2bf(float f){
  union { float f; unsigned u; } x; x.f = f;
  unsigned u = x.u;
  unsigned r = (u + 0x7FFFu + ((u >> 16) & 1u)) >> 16;   // RNE
  return (unsigned short)r;
}
#if __has_builtin(__builtin_amdgcn_cvt_pk_bf16_f32)
__device__ __forceinline__ unsigned pkbf(float a, float b){
  union { v2bf v; unsigned u; } x;
  x.v = __builtin_amdgcn_cvt_pk_bf16_f32(a, b);          // 1 VALU op
  return x.u;
}
#else
__device__ __forceinline__ unsigned pkbf(float a, float b){
  return (unsigned)f2bf(a) | ((unsigned)f2bf(b) << 16);
}
#endif
// fast sigmoid: v_exp + v_rcp (no IEEE div chains)
__device__ __forceinline__ float sigmoidf(float x){
  return __builtin_amdgcn_rcpf(1.0f + __expf(-x));
}

// async copy 16B global -> LDS (GV mode), tracked by ASYNCcnt
__device__ __forceinline__ void async_g2l_b128(unsigned lds_addr, const void* gptr){
  asm volatile("global_load_async_to_lds_b128 %0, %1, off"
               :: "v"(lds_addr), "v"(gptr) : "memory");
}
__device__ __forceinline__ void wait_async0(){
  asm volatile("s_wait_asynccnt 0x0" ::: "memory");
}

// ---------------------------------------------------------------------------
// k2: per-channel NT GEMM out_c = A_c(768x768) * B_c^T, bf16 WMMA
//     block tile 128x128; double-buffered 128x32 K-slabs staged with
//     GLOBAL_LOAD_ASYNC_TO_LDS_B128 (ASYNCcnt) overlapping WMMA compute
// ---------------------------------------------------------------------------
__global__ __launch_bounds__(256) void k2_einsum(
    const unsigned short* __restrict__ a_bf,
    const unsigned short* __restrict__ b_bf,
    float* __restrict__ out_e)
{
  __shared__ unsigned short lA[2][128 * 32];   // 2 x 8 KB
  __shared__ unsigned short lB[2][128 * 32];   // 2 x 8 KB
  const int tid  = threadIdx.x;
  const int c    = blockIdx.z;
  const int i0   = blockIdx.x * 128;
  const int j0   = blockIdx.y * 128;
  const size_t base = (size_t)c * NN;
  const int wave = tid >> 5, lane = tid & 31;
  const int lrow = lane & 15, lhi = lane >> 4;
  const int wm = wave & 1, wn = wave >> 1;     // wave grid 2(m) x 4(n)
  v8f acc[4][2] = {};

  const int r0a = tid >> 1;                    // staging: 2 threads / row
  const int ch0 = (tid & 1) * 2;               // 16B chunk pair

  const unsigned short* gA = a_bf + base + (size_t)(i0 + r0a)*N_DIM + ch0*8;
  const unsigned short* gB = b_bf + base + (size_t)(j0 + r0a)*N_DIM + ch0*8;
  const int lo = r0a*32 + ch0*8;               // element offset in slab

  auto stage = [&](int k0, int buf){
    async_g2l_b128((unsigned)(size_t)&lA[buf][lo],     gA + k0);
    async_g2l_b128((unsigned)(size_t)&lA[buf][lo + 8], gA + k0 + 8);
    async_g2l_b128((unsigned)(size_t)&lB[buf][lo],     gB + k0);
    async_g2l_b128((unsigned)(size_t)&lB[buf][lo + 8], gB + k0 + 8);
  };

  stage(0, 0);
  wait_async0();
  __syncthreads();

  for (int ks = 0; ks < N_DIM/32; ++ks){
    const int buf = ks & 1;
    if (ks + 1 < N_DIM/32) stage((ks + 1) * 32, buf ^ 1);   // overlap with WMMA

    FragA fb[2];
    #pragma unroll
    for (int nt = 0; nt < 2; ++nt){
      const unsigned short* bp = &lB[buf][(wn*32 + nt*16 + lrow)*32 + lhi*16];
      fb[nt].q[0] = *(const uint4*)bp;
      fb[nt].q[1] = *(const uint4*)(bp + 8);
    }
    #pragma unroll
    for (int mtile = 0; mtile < 4; ++mtile){
      FragA fa;
      const unsigned short* ap = &lA[buf][(wm*64 + mtile*16 + lrow)*32 + lhi*16];
      fa.q[0] = *(const uint4*)ap;
      fa.q[1] = *(const uint4*)(ap + 8);
      #pragma unroll
      for (int nt = 0; nt < 2; ++nt)
        acc[mtile][nt] = __builtin_amdgcn_wmma_f32_16x16x32_bf16(
            false, fa.v, false, fb[nt].v, (short)0, acc[mtile][nt], false, false);
    }
    wait_async0();                             // next slab landed in LDS
    __syncthreads();
  }

  float* outc = out_e + base;
  #pragma unroll
  for (int mtile = 0; mtile < 4; ++mtile){
    #pragma unroll
    for (int nt = 0; nt < 2; ++nt){
      int gj = j0 + wn*32 + nt*16 + lrow;
      int gm = i0 + wm*64 + mtile*16 + lhi*8;
      #pragma unroll
      for (int v = 0; v < 8; ++v)
        outc[(size_t)(gm + v)*N_DIM + gj] = acc[mtile][nt][v];
    }
  }
}

// ---------------------------------------------------------------------------
// k1: LayerNorm 64 rows -> bf16 in LDS -> 64x640x128 bf16 WMMA GEMM vs wb
//     -> pg = proj*sigmoid(gate) de-interleaved to a_bf/b_bf in (c,i,k)
//     -> g_s = sigmoid(x @ w_glin^T) in (p,c)
// ---------------------------------------------------------------------------
__global__ __launch_bounds__(256) void k1_ln_proj(
    const float* __restrict__ pair,
    const float* __restrict__ lnw, const float* __restrict__ lnb,
    const unsigned short* __restrict__ wb,
    unsigned short* __restrict__ a_bf, unsigned short* __restrict__ b_bf,
    float* __restrict__ g_s)
{
  __shared__ unsigned short xs[64 * 128];      // 16 KB bf16 layernormed rows
  const int tid   = threadIdx.x;
  const int p0    = blockIdx.x * 64;           // 64 consecutive (i,k) positions
  const int i_row = p0 / N_DIM;
  const int kbase = p0 % N_DIM;

  // Phase A: layernorm, 4 lanes per row (contiguous within a wave32)
  {
    int pos = tid >> 2, sub = tid & 3;
    const float* src = pair + (size_t)(p0 + pos) * C_DIM + sub * 32;
    float vbuf[32];
    float s = 0.f, s2 = 0.f;
    #pragma unroll
    for (int j = 0; j < 8; ++j){
      float4 f = ((const float4*)src)[j];
      vbuf[4*j+0]=f.x; vbuf[4*j+1]=f.y; vbuf[4*j+2]=f.z; vbuf[4*j+3]=f.w;
      s  += f.x + f.y + f.z + f.w;
      s2 += f.x*f.x + f.y*f.y + f.z*f.z + f.w*f.w;
    }
    s  += __shfl_xor(s, 1);  s  += __shfl_xor(s, 2);
    s2 += __shfl_xor(s2, 1); s2 += __shfl_xor(s2, 2);
    float mu   = s * (1.f/128.f);
    float var  = s2 * (1.f/128.f) - mu * mu;
    float rstd = rsqrtf(var + 1e-5f);
    #pragma unroll
    for (int j = 0; j < 32; j += 2){
      int c = sub * 32 + j;
      float z0 = (vbuf[j]   - mu) * rstd * lnw[c]   + lnb[c];
      float z1 = (vbuf[j+1] - mu) * rstd * lnw[c+1] + lnb[c+1];
      *(unsigned*)&xs[pos*128 + c] = pkbf(z0, z1);       // packed b32 ds_store
    }
  }
  __syncthreads();

  // Phase B: 96 tile-units (4 m-tiles x 24 n-units), 12 per wave
  const int wave = tid >> 5, lane = tid & 31;
  const int lrow = lane & 15, lhi = lane >> 4;

  for (int iter = 0; iter < 12; ++iter){
    int u  = iter * 8 + wave;
    int mt = u & 3;
    int nu = u >> 2;                           // 0..23
    if (nu < 16){                              // paired proj/gate 16-col tiles
      int n0p = nu * 16;
      int n0g = 256 + nu * 16;
      v8f accP = {}; v8f accG = {};
      #pragma unroll
      for (int kk = 0; kk < 4; ++kk){
        FragA fa, fbp, fbg;
        const unsigned short* ap = xs + (mt*16 + lrow)*128 + kk*32 + lhi*16;
        fa.q[0] = *(const uint4*)ap;
        fa.q[1] = *(const uint4*)(ap + 8);
        const unsigned short* bp = wb + (size_t)(n0p + lrow)*128 + kk*32 + lhi*16;
        fbp.q[0] = *(const uint4*)bp;
        fbp.q[1] = *(const uint4*)(bp + 8);
        const unsigned short* bg = wb + (size_t)(n0g + lrow)*128 + kk*32 + lhi*16;
        fbg.q[0] = *(const uint4*)bg;
        fbg.q[1] = *(const uint4*)(bg + 8);
        accP = __builtin_amdgcn_wmma_f32_16x16x32_bf16(false, fa.v, false, fbp.v,
                                                       (short)0, accP, false, false);
        accG = __builtin_amdgcn_wmma_f32_16x16x32_bf16(false, fa.v, false, fbg.v,
                                                       (short)0, accG, false, false);
      }
      float pg[8];
      #pragma unroll
      for (int v = 0; v < 8; ++v) pg[v] = accP[v] * sigmoidf(accG[v]);
      int col = n0p + lrow;                    // 0..255
      int ch  = col >> 1;
      unsigned short* dst = ((col & 1) == 0) ? a_bf : b_bf;
      size_t off = (size_t)ch*NN + (size_t)i_row*N_DIM + kbase + mt*16 + lhi*8;
      uint4 q;
      q.x = pkbf(pg[0], pg[1]); q.y = pkbf(pg[2], pg[3]);
      q.z = pkbf(pg[4], pg[5]); q.w = pkbf(pg[6], pg[7]);
      *(uint4*)(dst + off) = q;                // contiguous in k -> b128 store
    } else {                                   // glin -> output gate
      int n0 = 512 + (nu - 16) * 16;
      v8f acc = {};
      #pragma unroll
      for (int kk = 0; kk < 4; ++kk){
        FragA fa, fb;
        const unsigned short* ap = xs + (mt*16 + lrow)*128 + kk*32 + lhi*16;
        fa.q[0] = *(const uint4*)ap;
        fa.q[1] = *(const uint4*)(ap + 8);
        const unsigned short* bp = wb + (size_t)(n0 + lrow)*128 + kk*32 + lhi*16;
        fb.q[0] = *(const uint4*)bp;
        fb.q[1] = *(const uint4*)(bp + 8);
        acc = __builtin_amdgcn_wmma_f32_16x16x32_bf16(false, fa.v, false, fb.v,
                                                      (short)0, acc, false, false);
      }
      int c = (n0 - 512) + lrow;
      #pragma unroll
      for (int v = 0; v < 8; ++v){
        int p = p0 + mt*16 + lhi*8 + v;
        g_s[(size_t)p*C_DIM + c] = sigmoidf(acc[v]);
      }
    }
  }
}

// ---------------------------------------------------------------------------
// k3: gather (c,p)->LDS, LayerNorm over c, 128x128x128 bf16 WMMA vs w_out,
//     multiply by stored sigmoid gate, store fp32 (p,c)
// ---------------------------------------------------------------------------
__global__ __launch_bounds__(256) void k3_out(
    const float* __restrict__ out_e,
    const float* __restrict__ lnw, const float* __restrict__ lnb,
    const unsigned short* __restrict__ wb,     // rows 640..767 = w_out bf16
    const float* __restrict__ g_s,
    float* __restrict__ out)
{
  __shared__ float          ys[128 * 130];     // padded: no bank conflicts
  __shared__ unsigned short zs[128 * 136];     // bf16 normalized tile
  const int tid = threadIdx.x;
  const int p0  = blockIdx.x * 128;

  for (int idx = tid; idx < 128 * 128; idx += 256){
    int cc = idx >> 7, pp = idx & 127;
    ys[pp*130 + cc] = out_e[(size_t)cc*NN + p0 + pp];   // coalesced per channel
  }
  __syncthreads();

  if (tid < 128){
    const float* row = ys + tid * 130;
    float s = 0.f, s2 = 0.f;
    #pragma unroll
    for (int cI = 0; cI < 128; ++cI){ float v = row[cI]; s += v; s2 += v*v; }
    float mu   = s * (1.f/128.f);
    float rstd = rsqrtf(s2 * (1.f/128.f) - mu*mu + 1e-5f);
    unsigned short* zr = zs + tid * 136;
    #pragma unroll
    for (int cI = 0; cI < 128; cI += 2){
      float z0 = (row[cI]   - mu) * rstd * lnw[cI]   + lnb[cI];
      float z1 = (row[cI+1] - mu) * rstd * lnw[cI+1] + lnb[cI+1];
      *(unsigned*)&zr[cI] = pkbf(z0, z1);                // packed b32 ds_store
    }
  }
  __syncthreads();

  const int wave = tid >> 5, lane = tid & 31;
  const int lrow = lane & 15, lhi = lane >> 4;
  for (int q = 0; q < 8; ++q){
    int tile = q * 8 + wave;                   // 64 tiles over 8 waves
    int mt = tile >> 3, nt = tile & 7;
    v8f acc = {};
    #pragma unroll
    for (int kk = 0; kk < 4; ++kk){
      FragA fa, fb;
      const unsigned short* ap = zs + (mt*16 + lrow)*136 + kk*32 + lhi*16;
      fa.q[0] = *(const uint4*)ap;
      fa.q[1] = *(const uint4*)(ap + 8);
      const unsigned short* bp = wb + (size_t)(640 + nt*16 + lrow)*128 + kk*32 + lhi*16;
      fb.q[0] = *(const uint4*)bp;
      fb.q[1] = *(const uint4*)(bp + 8);
      acc = __builtin_amdgcn_wmma_f32_16x16x32_bf16(false, fa.v, false, fb.v,
                                                    (short)0, acc, false, false);
    }
    int cc = nt*16 + lrow;
    #pragma unroll
    for (int v = 0; v < 8; ++v){
      int p = p0 + mt*16 + lhi*8 + v;
      size_t o = (size_t)p*C_DIM + cc;
      out[o] = acc[v] * g_s[o];
    }
  }
}

// ---------------------------------------------------------------------------
// k0: pack [w_proj(256);w_gate(256);w_glin(128);w_out(128)] x 128 into bf16
// ---------------------------------------------------------------------------
__global__ __launch_bounds__(256) void k0_pack_weights(
    const float* __restrict__ wp, const float* __restrict__ wg,
    const float* __restrict__ wl, const float* __restrict__ wo,
    unsigned short* __restrict__ wb)
{
  int idx = blockIdx.x * 256 + threadIdx.x;
  if (idx >= 768 * 128) return;
  int row = idx >> 7, col = idx & 127;
  float v;
  if      (row < 256) v = wp[ row       *128 + col];
  else if (row < 512) v = wg[(row - 256)*128 + col];
  else if (row < 640) v = wl[(row - 512)*128 + col];
  else                v = wo[(row - 640)*128 + col];
  wb[idx] = f2bf(v);
}

// ---------------------------------------------------------------------------
extern "C" void kernel_launch(void* const* d_in, const int* in_sizes, int n_in,
                              void* d_out, int out_size, void* d_ws, size_t ws_size,
                              hipStream_t stream) {
  const float* pair    = (const float*)d_in[0];
  const float* ln_in_w = (const float*)d_in[1];
  const float* ln_in_b = (const float*)d_in[2];
  const float* w_proj  = (const float*)d_in[3];
  const float* w_gate  = (const float*)d_in[4];
  const float* ln_c_w  = (const float*)d_in[5];
  const float* ln_c_b  = (const float*)d_in[6];
  const float* w_out   = (const float*)d_in[7];
  const float* w_glin  = (const float*)d_in[8];

  char* ws = (char*)d_ws;
  const size_t MB = 1ull << 20;
  unsigned short* wb   = (unsigned short*)(ws);                        // 192 KB
  unsigned short* a_bf = (unsigned short*)(ws + 1*MB);                 // 144 MB
  unsigned short* b_bf = (unsigned short*)(ws + 1*MB + 151*MB);       // 144 MB
  float*          g_s  = (float*)         (ws + 1*MB + 302*MB);       // 288 MB
  float*          oute = (float*)         (ws + 1*MB + 302*MB + 289*MB); // 288 MB

  k0_pack_weights<<<384, 256, 0, stream>>>(w_proj, w_gate, w_glin, w_out, wb);
  k1_ln_proj<<<(768*768)/64, 256, 0, stream>>>(pair, ln_in_w, ln_in_b, wb,
                                               a_bf, b_bf, g_s);
  dim3 g2(6, 6, 128);
  k2_einsum<<<g2, 256, 0, stream>>>(a_bf, b_bf, oute);
  k3_out<<<(768*768)/128, 256, 0, stream>>>(oute, ln_c_w, ln_c_b, wb, g_s,
                                            (float*)d_out);
}
